// LinearSegmentMemoryWrapper_6210522710455
// MI455X (gfx1250) — matrix operand order repaired and checked by reading, not compiled, not run on hardware
//
#include <hip/hip_runtime.h>
#include <hip/hip_bf16.h>
#include <math.h>

#define B_SZ   8
#define S_SZ   2048
#define D_SZ   512
#define H_SZ   512
#define K_SZ   512
#define DECAY_F 0.97f
#define CHUNK  32     // tokens per scan chunk
#define HT     64     // h rows per scan workgroup

typedef __attribute__((ext_vector_type(16))) __bf16 v16bf;
typedef __attribute__((ext_vector_type(8)))  float  v8f;

union BFr { v16bf v; unsigned short u[16]; };

__device__ __forceinline__ unsigned short f2bf(float f) {
  unsigned u = __builtin_bit_cast(unsigned, f);
  u += 0x7FFFu + ((u >> 16) & 1u);            // round-to-nearest-even
  return (unsigned short)(u >> 16);
}
__device__ __forceinline__ float bf2f(unsigned short h) {
  unsigned u = ((unsigned)h) << 16;
  return __builtin_bit_cast(float, u);
}

__device__ __forceinline__ v8f wmma_bf16(v16bf a, v16bf b, v8f c) {
  // D(f32 16x16) = A(bf16 16x32) * B(bf16 32x16) + C
  return __builtin_amdgcn_wmma_f32_16x16x32_bf16(false, a, false, b, (short)0, c,
                                                 false, false);
}

// ---- fragment assembly from LDS (bf16 as ushort), contiguous k order, k = 16*hi + e, in A and B alike
// A matrix 16x32: lane m = lane&15; per-lane elements are one contiguous 16-run, k = 16*hi + e.
__device__ __forceinline__ v16bf load_a(const unsigned short* base, int ld, int r0, int c0) {
  int lane = threadIdx.x & 31;
  int m = lane & 15, hi = lane >> 4;
  BFr f;
#pragma unroll
  for (int e = 0; e < 16; ++e) {
    int v = e >> 1, p = e & 1;
    int k = (hi << 4) + e;
    f.u[e] = base[(size_t)(r0 + m) * ld + c0 + k];
  }
  return f.v;
}
// B matrix 32x16 from K-major storage (strided; used only for state update)
__device__ __forceinline__ v16bf load_b(const unsigned short* base, int ld, int k0, int n0) {
  int lane = threadIdx.x & 31;
  int n = lane & 15, hi = lane >> 4;
  BFr f;
#pragma unroll
  for (int e = 0; e < 16; ++e) {
    int k = (hi << 4) + e;
    f.u[e] = base[(size_t)(k0 + k) * ld + n0 + n];
  }
  return f.v;
}
// B matrix 32x16 from N-major storage: B[k][n] = base[(n0+n)*ld + k0+k]
// -> 16 contiguous ushorts per lane -> ds_load_b128 pairs
__device__ __forceinline__ v16bf load_bt(const unsigned short* base, int ld, int k0, int n0) {
  int lane = threadIdx.x & 31;
  int n = lane & 15, hi = lane >> 4;
  BFr f;
#pragma unroll
  for (int e = 0; e < 16; ++e) {
    int k = (hi << 4) + e;
    f.u[e] = base[(size_t)(n0 + n) * ld + k0 + k];
  }
  return f.v;
}

// ------------------------------------------------- TDM (Tensor Data Mover)
#if __has_builtin(__builtin_amdgcn_tensor_load_to_lds)
#define USE_TDM 1
#else
#define USE_TDM 0
#endif
#if __has_include(<hip/amd_detail/amd_gfx1250_TDM.h>)
#define TDM_SIX_ARGS 1
#else
#define TDM_SIX_ARGS 0
#endif

#if USE_TDM
typedef __attribute__((ext_vector_type(4))) unsigned int u32x4;
typedef __attribute__((ext_vector_type(8))) int          i32x8;
typedef __attribute__((ext_vector_type(4))) int          i32x4;

// 2D tile load Global->LDS with LDS row padding of 4 DWORDs per 256 DWORDs
// (bf16 row of 512 elems = 1024B -> LDS row stride 520 ushorts). ISA ch.8 D#.
__device__ __forceinline__ void tdm_load_tile_2d(const void* gsrc, unsigned ldsOff,
                                                 unsigned rows, unsigned cols,
                                                 unsigned strideElems) {
  unsigned long long ga = (unsigned long long)gsrc;
  u32x4 g0;
  g0.x = 1u;                                             // count=1, load, no gather
  g0.y = ldsOff;                                         // lds_addr (bytes)
  g0.z = (unsigned)ga;                                   // global_addr[31:0]
  g0.w = (unsigned)((ga >> 32) & 0x01FFFFFFu) | (2u << 30);  // ga[56:32] | type=2
  i32x8 g1;
  g1[0] = (int)((1u << 16)      // data_size = 2 bytes
              | (1u << 20)      // pad_enable
              | (7u << 22)      // pad_interval: 256 DWORDs (one 1024B row)
              | (3u << 25));    // pad_amount: 4 DWORDs (16B)
  g1[1] = (int)((cols & 0xFFFFu) << 16);                         // tensor_dim0 lo
  g1[2] = (int)(((cols >> 16) & 0xFFFFu) | ((rows & 0xFFFFu) << 16)); // dim0 hi|dim1 lo
  g1[3] = (int)(((rows >> 16) & 0xFFFFu) | ((cols & 0xFFFFu) << 16)); // dim1 hi|tile_dim0
  g1[4] = (int)(rows & 0xFFFFu);                                 // tile_dim1, tile_dim2=0
  g1[5] = (int)strideElems;                                      // tensor_dim0_stride lo
  g1[6] = 0; g1[7] = 0;
  i32x4 gz = (i32x4)0;
#if TDM_SIX_ARGS
  i32x8 gz8 = (i32x8)0;
  __builtin_amdgcn_tensor_load_to_lds(g0, g1, gz, gz, gz8, 0);
#else
  __builtin_amdgcn_tensor_load_to_lds(g0, g1, gz, gz, 0);
#endif
}
#endif

// ---------------------------------------------------------------- cvt f32->bf16
__global__ void cvt_bf16_kernel(const float* __restrict__ src,
                                unsigned short* __restrict__ dst, int n4) {
  int i = blockIdx.x * blockDim.x + threadIdx.x;
  if (i < n4) {
    float4 f = ((const float4*)src)[i];
    uint2 o;
    o.x = (unsigned)f2bf(f.x) | ((unsigned)f2bf(f.y) << 16);
    o.y = (unsigned)f2bf(f.z) | ((unsigned)f2bf(f.w) << 16);
    ((uint2*)dst)[i] = o;
  }
}

// ---------------------------------------------------------- generic 512-K GEMM
// out[rows, 512] = A[rows,512](bf16) @ W^T + bias, W row-major [512 x 512].
// do_norm: l2-normalize each output row. Else scale by out_scale.
__global__ __launch_bounds__(256)
void gemm512_kernel(const unsigned short* __restrict__ A,
                    const unsigned short* __restrict__ W,
                    const float* __restrict__ bias,
                    unsigned short* __restrict__ dst16,
                    float* __restrict__ dst32,
                    int do_norm, float out_scale) {
  __shared__ unsigned short xs[64][520];    // 64 rows of A, full K=512
  __shared__ unsigned short wtT[512][48];   // W^T K-slice, N-major: wtT[n][d]
  __shared__ float rowsum[64];

  const int tid = threadIdx.x;
  const int wv = tid >> 5;
  const int lane = tid & 31, ln = lane & 15, hi = lane >> 4;
  const int rows0 = blockIdx.x * 64;

  for (int idx = tid; idx < 64 * 64; idx += 256) {       // 128-bit staging
    int r = idx >> 6, c8 = (idx & 63) << 3;
    *(uint4*)&xs[r][c8] = *(const uint4*)(A + (size_t)(rows0 + r) * 512 + c8);
  }
  if (tid < 64) rowsum[tid] = 0.f;
  __syncthreads();

  v8f acc[4][4] = {};   // [i16][t] : rows i16*16.., cols (wv*4+t)*16..

  for (int kk = 0; kk < 512; kk += 32) {
    for (int idx = tid; idx < 512 * 4; idx += 256) {     // 128-bit staging
      int n = idx >> 2, d8 = (idx & 3) << 3;
      *(uint4*)&wtT[n][d8] = *(const uint4*)(W + (size_t)n * 512 + kk + d8);
    }
    __syncthreads();
    v16bf bfr[4];
#pragma unroll
    for (int t = 0; t < 4; ++t) bfr[t] = load_bt(&wtT[0][0], 48, 0, (wv * 4 + t) * 16);
#pragma unroll
    for (int i16 = 0; i16 < 4; ++i16) {
      v16bf af = load_a(&xs[0][0], 520, i16 * 16, kk);
#pragma unroll
      for (int t = 0; t < 4; ++t) acc[i16][t] = wmma_bf16(af, bfr[t], acc[i16][t]);
    }
    __syncthreads();
  }

  // epilogue: bias, optional row l2-norm (cross-wave via LDS), store
  float bvals[4];
#pragma unroll
  for (int t = 0; t < 4; ++t) bvals[t] = bias[(wv * 4 + t) * 16 + ln];
#pragma unroll
  for (int i16 = 0; i16 < 4; ++i16)
#pragma unroll
    for (int t = 0; t < 4; ++t)
#pragma unroll
      for (int r = 0; r < 8; ++r) acc[i16][t][r] += bvals[t];

  if (do_norm) {
#pragma unroll
    for (int i16 = 0; i16 < 4; ++i16) {
#pragma unroll
      for (int r = 0; r < 8; ++r) {
        float ss = 0.f;
#pragma unroll
        for (int t = 0; t < 4; ++t) { float v = acc[i16][t][r]; ss += v * v; }
        ss += __shfl_xor(ss, 1, 32);
        ss += __shfl_xor(ss, 2, 32);
        ss += __shfl_xor(ss, 4, 32);
        ss += __shfl_xor(ss, 8, 32);
        if (ln == 0) atomicAdd(&rowsum[i16 * 16 + hi * 8 + r], ss);
      }
    }
    __syncthreads();
  }
#pragma unroll
  for (int i16 = 0; i16 < 4; ++i16) {
#pragma unroll
    for (int r = 0; r < 8; ++r) {
      int row_l = i16 * 16 + hi * 8 + r;
      float mult = do_norm ? (1.0f / fmaxf(sqrtf(rowsum[row_l]), 1e-6f)) : out_scale;
#pragma unroll
      for (int t = 0; t < 4; ++t) {
        float v = acc[i16][t][r] * mult;
        size_t g = (size_t)(rows0 + row_l) * 512 + (wv * 4 + t) * 16 + ln;
        if (dst32) dst32[g] = v; else dst16[g] = f2bf(v);
      }
    }
  }
}

// ------------------------------------------------------- chunked decay scan
// One workgroup = (batch b, 64-row h-slice). f32 state M[64,512] lives in WMMA
// accumulators (16 tiles per wave, wave grid 4(h) x 2(k-half)). 64 chunk steps.
// Per chunk:  sA = mask(Qc Kc^T);  hid = sA V + d^{i+1} Qc M^T;  M = d^C M + Vw Kc.
__global__ __launch_bounds__(256)
void scan_kernel(const unsigned short* __restrict__ kb,
                 const unsigned short* __restrict__ vb,
                 const unsigned short* __restrict__ qb,
                 unsigned short* __restrict__ hb) {
  __shared__ unsigned short Kc[CHUNK][520];   // chunk keys   [32 x 512]
  __shared__ unsigned short Qc[CHUNK][520];   // chunk queries[32 x 512]
  __shared__ unsigned short bM[HT][520];      // bf16 snapshot of state M [64 x 512]
  __shared__ unsigned short VsT[HT][48];      // value slice, N-major [64h x 32j]
  __shared__ unsigned short Vw[HT][48];       // decay-weighted V^T [64h x 32j]
  __shared__ unsigned short sA[CHUNK][48];    // masked scores [32 x 32]
  __shared__ float powtab[40];                // DECAY^i, i=0..33

  const int tid = threadIdx.x;
  const int wv = tid >> 5;
  const int lane = tid & 31, ln = lane & 15, hi = lane >> 4;
  const int b = blockIdx.x >> 3;
  const int ht = blockIdx.x & 7;
  const int wh = wv >> 1, wk = wv & 1;   // state tile ownership

  if (tid < 34) powtab[tid] = __powf(DECAY_F, (float)tid);

#if USE_TDM
  const unsigned ldsKc = (unsigned)(unsigned long long)&Kc[0][0];
  const unsigned ldsQc = (unsigned)(unsigned long long)&Qc[0][0];
#endif

  v8f accM[16];
#pragma unroll
  for (int i = 0; i < 16; ++i) accM[i] = (v8f){};
  __syncthreads();

  for (int c = 0; c < S_SZ / CHUNK; ++c) {
    const int s0 = c * CHUNK;

    // ---- stage K/Q chunk tiles: async TDM DMA (wave 0) or manual 128-bit path
#if USE_TDM
    if (wv == 0) {
      tdm_load_tile_2d(kb + ((size_t)b * S_SZ + s0) * 512, ldsKc, CHUNK, 512, 512);
      tdm_load_tile_2d(qb + ((size_t)b * S_SZ + s0) * 512, ldsQc, CHUNK, 512, 512);
    }
#else
    for (int idx = tid; idx < CHUNK * 64; idx += 256) {
      int j = idx >> 6, k8 = (idx & 63) << 3;
      size_t g = ((size_t)b * S_SZ + s0 + j) * 512 + k8;
      *(uint4*)&Kc[j][k8] = *(const uint4*)(kb + g);
      *(uint4*)&Qc[j][k8] = *(const uint4*)(qb + g);
    }
#endif
    // ---- value slice (N-major) + decay-weighted transpose, overlapped with DMA
    for (int idx = tid; idx < CHUNK * 8; idx += 256) {
      int j = idx >> 3, h8 = (idx & 7) << 3;
      uint4 raw = *(const uint4*)(vb + ((size_t)b * S_SZ + s0 + j) * 512 + ht * HT + h8);
      const unsigned short* u = (const unsigned short*)&raw;
#pragma unroll
      for (int t = 0; t < 8; ++t) {
        VsT[h8 + t][j] = u[t];
        Vw[h8 + t][j] = f2bf(bf2f(u[t]) * powtab[CHUNK - 1 - j]);  // d^{C-1-j} v'
      }
    }
    // ---- bf16 snapshot of state M (read by the inter-chunk GEMM)
#pragma unroll
    for (int kt = 0; kt < 16; ++kt) {
      int kbase = wk * 256 + kt * 16;
#pragma unroll
      for (int r = 0; r < 8; ++r)
        bM[wh * 16 + hi * 8 + r][kbase + ln] = f2bf(accM[kt][r]);
    }
#if USE_TDM
    if (wv == 0) __builtin_amdgcn_s_wait_tensorcnt((short)0);
#endif
    __syncthreads();

    // ---- scores: sA[i][j] = d^{i-j} * (q_i . k_j), lower-triangular (waves 0-3)
    if (wv < 4) {
      int i16 = wv >> 1, j16 = wv & 1;
      v8f s = {};
      for (int kk = 0; kk < 512; kk += 32) {
        v16bf a  = load_a (&Qc[0][0], 520, i16 * 16, kk);
        v16bf bb = load_bt(&Kc[0][0], 520, kk, j16 * 16);   // K^T, contiguous
        s = wmma_bf16(a, bb, s);
      }
#pragma unroll
      for (int r = 0; r < 8; ++r) {
        int i = i16 * 16 + hi * 8 + r, j = j16 * 16 + ln;
        float wgt = (j <= i) ? powtab[i - j] : 0.f;
        sA[i][j] = f2bf(s[r] * wgt);
      }
    }
    __syncthreads();

    // ---- hidden: intra (sA @ V) + d^{i+1} * inter (Q @ M^T); one tile per wave
    {
      int i16 = wv >> 2, n16 = wv & 3;
      v8f aI = {};
      {
        v16bf a  = load_a (&sA[0][0], 48, i16 * 16, 0);
        v16bf bb = load_bt(&VsT[0][0], 48, 0, n16 * 16);
        aI = wmma_bf16(a, bb, aI);
      }
      v8f aE = {};
      for (int kk = 0; kk < 512; kk += 32) {
        v16bf a  = load_a (&Qc[0][0], 520, i16 * 16, kk);
        v16bf bb = load_bt(&bM[0][0], 520, kk, n16 * 16);   // M^T, contiguous
        aE = wmma_bf16(a, bb, aE);
      }
#pragma unroll
      for (int r = 0; r < 8; ++r) {
        int i_l = i16 * 16 + hi * 8 + r;
        float v = aI[r] + powtab[i_l + 1] * aE[r];
        size_t g = ((size_t)b * S_SZ + s0 + i_l) * 512 + ht * HT + n16 * 16 + ln;
        hb[g] = f2bf(v);
      }
    }

    // ---- state update: M = d^C * M + Vw @ Kc (f32 accumulators, per-wave tiles)
    {
      const float dC = powtab[CHUNK];
      v16bf a = load_a(&Vw[0][0], 48, wh * 16, 0);
#pragma unroll
      for (int kt = 0; kt < 16; ++kt) {
        int kbase = wk * 256 + kt * 16;
#pragma unroll
        for (int r = 0; r < 8; ++r) accM[kt][r] *= dC;
        v16bf bb = load_b(&Kc[0][0], 520, 0, kbase);
        accM[kt] = wmma_bf16(a, bb, accM[kt]);
      }
    }
    __syncthreads();   // protect LDS before next chunk's staging
  }
}

// --------------------------------------------------------------------- driver
extern "C" void kernel_launch(void* const* d_in, const int* in_sizes, int n_in,
                              void* d_out, int out_size, void* d_ws, size_t ws_size,
                              hipStream_t stream) {
  (void)in_sizes; (void)n_in; (void)out_size; (void)ws_size;
  const float* x  = (const float*)d_in[0];
  const float* Wk = (const float*)d_in[1]; const float* bk = (const float*)d_in[2];
  const float* Wv = (const float*)d_in[3]; const float* bv = (const float*)d_in[4];
  const float* Wq = (const float*)d_in[5]; const float* bq = (const float*)d_in[6];
  const float* Wo = (const float*)d_in[7]; const float* bo = (const float*)d_in[8];
  float* out = (float*)d_out;

  char* ws = (char*)d_ws;
  size_t off = 0;
  auto take = [&](size_t bytes) -> unsigned short* {
    unsigned short* p = (unsigned short*)(ws + off);
    off += (bytes + 255) & ~(size_t)255;
    return p;
  };
  const size_t NROW = (size_t)B_SZ * S_SZ;                  // 16384
  unsigned short* xb  = take(NROW * D_SZ * 2);
  unsigned short* wkb = take((size_t)K_SZ * D_SZ * 2);
  unsigned short* wvb = take((size_t)H_SZ * D_SZ * 2);
  unsigned short* wqb = take((size_t)K_SZ * D_SZ * 2);
  unsigned short* wob = take((size_t)D_SZ * H_SZ * 2);
  unsigned short* kb  = take(NROW * K_SZ * 2);
  unsigned short* vb  = take(NROW * H_SZ * 2);
  unsigned short* qb  = take(NROW * K_SZ * 2);
  unsigned short* hb  = take(NROW * H_SZ * 2);

  int ne4 = (int)(NROW * D_SZ) / 4;
  cvt_bf16_kernel<<<(ne4 + 255) / 256, 256, 0, stream>>>(x, xb, ne4);
  int nw4 = (K_SZ * D_SZ) / 4;
  cvt_bf16_kernel<<<(nw4 + 255) / 256, 256, 0, stream>>>(Wk, wkb, nw4);
  cvt_bf16_kernel<<<(nw4 + 255) / 256, 256, 0, stream>>>(Wv, wvb, nw4);
  cvt_bf16_kernel<<<(nw4 + 255) / 256, 256, 0, stream>>>(Wq, wqb, nw4);
  cvt_bf16_kernel<<<(nw4 + 255) / 256, 256, 0, stream>>>(Wo, wob, nw4);

  const float scale = (1.0f - DECAY_F) / sqrtf((float)K_SZ);  // folded into v'
  unsigned gblk = (unsigned)(NROW / 64);
  gemm512_kernel<<<gblk, 256, 0, stream>>>(xb, wkb, bk, kb, nullptr, 1, 1.0f);
  gemm512_kernel<<<gblk, 256, 0, stream>>>(xb, wvb, bv, vb, nullptr, 0, scale);
  gemm512_kernel<<<gblk, 256, 0, stream>>>(xb, wqb, bq, qb, nullptr, 1, 1.0f);

  scan_kernel<<<B_SZ * (H_SZ / HT), 256, 0, stream>>>(kb, vb, qb, hb);

  gemm512_kernel<<<gblk, 256, 0, stream>>>(hb, wob, bo, nullptr, out, 0, 1.0f);
}
